// CharRNN_40604620816471
// MI455X (gfx1250) — matrix operand order repaired and checked by reading, not compiled, not run on hardware
//
#include <hip/hip_runtime.h>
#include <hip/hip_bf16.h>

// ---------------------------------------------------------------------------
// CharRNN forward for MI455X (gfx1250, wave32, WMMA).
//   hc     = hprev @ Whh + bh                      [2048]
//   h      = tanh(x @ Wxh + hc)                    [4096, 2048]  (bf16 in ws)
//   logits = h @ Why + by                          [4096, 32000] (f32 out)
// Output layout: logits (T*V floats) ++ h[T-1] (H floats).
// Both big GEMMs use v_wmma_f32_16x16x32_bf16 with f32 accumulation; f32
// inputs are converted to bf16 (RNE) during LDS staging so each global byte
// is read once. Memory floor ~1.6GB -> ~70us @ 23.3 TB/s; WMMA compute is
// far below that, so the kernels are bandwidth-bound by construction.
// gemm2's A tile (already bf16) is staged with global_load_async_to_lds_b128
// (ASYNCcnt path) instead of a VGPR round-trip.
// ---------------------------------------------------------------------------

typedef __attribute__((ext_vector_type(16))) __bf16 v16bf;
typedef __attribute__((ext_vector_type(8)))  float  v8f;

#define HIDDEN 2048
#define VOCAB  32000
#define TSEQ   4096

#define TM 128          // block tile M
#define TN 128          // block tile N
#define TK 32           // K step (one bf16 WMMA K)
#define LDSS 34         // LDS row stride in u16 (32 + 2 pad -> 17-bank step)
#define LDSSA2 40       // gemm2 A-tile stride in u16 (80B rows: 16B-aligned
                        // for async B128 LDS writes; 20-bank step, no conflict)

union FragAB { v16bf v; unsigned int u[8]; };
union FragC  { v8f   v; float        f[8]; };

__device__ __forceinline__ unsigned short f32_to_bf16_rne(float f) {
  unsigned int u = __float_as_uint(f);
  u += 0x7FFFu + ((u >> 16) & 1u);
  return (unsigned short)(u >> 16);
}

// ------------------------- hc = hprev @ Whh + bh ---------------------------
__global__ void __launch_bounds__(256)
hcontrib_kernel(const float* __restrict__ hprev, const float* __restrict__ Whh,
                const float* __restrict__ bh, float* __restrict__ hc) {
  const int j = blockIdx.x * 256 + threadIdx.x;      // 0..2047, coalesced
  float s = bh[j];
#pragma unroll 8
  for (int k = 0; k < HIDDEN; ++k)
    s = fmaf(hprev[k], Whh[(size_t)k * HIDDEN + j], s);
  hc[j] = s;
}

// -------- shared fragment-load helpers (CDNA5 ISA 7.12.2 layouts) ----------
__device__ __forceinline__ void load_a_frag(const unsigned short* As, int stride,
                                            int m, int lane, FragAB& a) {
  const int kb = (lane >> 4) ? 8 : 0;                // lanes>=16: K+8
#pragma unroll
  for (int v = 0; v < 8; ++v) {
    const int k = kb + ((v & 3) << 1) + ((v >= 4) ? 16 : 0);
    a.u[v] = *(const unsigned int*)&As[m * stride + k];
  }
}

__device__ __forceinline__ void load_b_frag(const unsigned short* Bs, int n,
                                            int lane, FragAB& b) {
  const int kb = (lane >> 4) ? 8 : 0;                // lanes>=16: K+8
#pragma unroll
  for (int v = 0; v < 8; ++v)
    b.u[v] = *(const unsigned int*)&Bs[n * LDSS + kb + ((v & 3) << 1) + ((v >= 4) ? 16 : 0)];
}

// ---------------- h = tanh(x @ Wxh + hc), store bf16 + last row ------------
__global__ void __launch_bounds__(256)
gemm1_tanh_kernel(const float* __restrict__ x, const float* __restrict__ Wxh,
                  const float* __restrict__ hc, unsigned short* __restrict__ hbf,
                  float* __restrict__ hlast_out) {
  __shared__ unsigned short As[TM * LDSS];
  __shared__ unsigned short Bs[TN * LDSS];

  const int tid  = threadIdx.x;
  const int lane = tid & 31;
  const int wave = tid >> 5;
  const int m0 = (wave & 3) * 32;                    // 4 waves along M
  const int n0 = (wave >> 2) * 64;                   // 2 waves along N
  const int blockN = blockIdx.x;                     // 0..15
  const int blockM = blockIdx.y;                     // 0..31

  FragC acc[2][4];
#pragma unroll
  for (int mi = 0; mi < 2; ++mi)
#pragma unroll
    for (int ni = 0; ni < 4; ++ni)
#pragma unroll
      for (int v = 0; v < 8; ++v) acc[mi][ni].f[v] = 0.0f;

  const int ar  = tid >> 1;                          // A stage: row 0..127
  const int akh = (tid & 1) * 16;                    //          k half
  const int bk  = tid >> 3;                          // B stage: k 0..31
  const int bn  = (tid & 7) * 16;                    //          n base
  const size_t arow = (size_t)(blockM * TM + ar);

  for (int kt = 0; kt < VOCAB; kt += TK) {
    { // stage A: x f32 -> bf16, row-major [128][32]
      const float* src = x + arow * VOCAB + kt + akh;
      unsigned int* dst = (unsigned int*)&As[ar * LDSS + akh];
#pragma unroll
      for (int i = 0; i < 16; i += 4) {
        float4 f = *(const float4*)(src + i);
        dst[(i >> 1) + 0] = (unsigned)f32_to_bf16_rne(f.x) |
                            ((unsigned)f32_to_bf16_rne(f.y) << 16);
        dst[(i >> 1) + 1] = (unsigned)f32_to_bf16_rne(f.z) |
                            ((unsigned)f32_to_bf16_rne(f.w) << 16);
      }
      __builtin_prefetch((const void*)(src + TK), 0, 0);  // next K tile
    }
    { // stage B: Wxh f32 -> bf16, transposed in LDS as [n][k]
      const float* src = Wxh + (size_t)(kt + bk) * HIDDEN + blockN * TN + bn;
#pragma unroll
      for (int i = 0; i < 16; i += 4) {
        float4 f = *(const float4*)(src + i);
        Bs[(bn + i + 0) * LDSS + bk] = f32_to_bf16_rne(f.x);
        Bs[(bn + i + 1) * LDSS + bk] = f32_to_bf16_rne(f.y);
        Bs[(bn + i + 2) * LDSS + bk] = f32_to_bf16_rne(f.z);
        Bs[(bn + i + 3) * LDSS + bk] = f32_to_bf16_rne(f.w);
      }
    }
    __syncthreads();

    FragAB a[2];
    const int mrow = lane & 15;
#pragma unroll
    for (int mi = 0; mi < 2; ++mi)
      load_a_frag(As, LDSS, m0 + mi * 16 + mrow, lane, a[mi]);
#pragma unroll
    for (int ni = 0; ni < 4; ++ni) {
      FragAB b;
      load_b_frag(Bs, n0 + ni * 16 + mrow, lane, b);
#pragma unroll
      for (int mi = 0; mi < 2; ++mi)
        acc[mi][ni].v = __builtin_amdgcn_wmma_f32_16x16x32_bf16(
            false, a[mi].v, false, b.v, (short)0, acc[mi][ni].v, false, false);
    }
    __syncthreads();
  }

  const int nlane = lane & 15;
  const int mhalf = (lane >> 4) ? 8 : 0;
#pragma unroll
  for (int mi = 0; mi < 2; ++mi)
#pragma unroll
    for (int ni = 0; ni < 4; ++ni) {
      const int col = blockN * TN + n0 + ni * 16 + nlane;
      const float hcv = hc[col];
#pragma unroll
      for (int v = 0; v < 8; ++v) {
        const int row = blockM * TM + m0 + mi * 16 + mhalf + v;
        const float hval = tanhf(acc[mi][ni].f[v] + hcv);
        hbf[(size_t)row * HIDDEN + col] = f32_to_bf16_rne(hval);
        if (row == TSEQ - 1) hlast_out[col] = hval;   // second output
      }
    }
}

// --------------------- logits = h_bf16 @ Why + by --------------------------
__global__ void __launch_bounds__(256)
gemm2_kernel(const unsigned short* __restrict__ hbf, const float* __restrict__ Why,
             const float* __restrict__ by, float* __restrict__ out) {
  __shared__ unsigned short As[TM * LDSSA2];   // 80B rows, 16B aligned
  __shared__ unsigned short Bs[TN * LDSS];

  const int tid  = threadIdx.x;
  const int lane = tid & 31;
  const int wave = tid >> 5;
  const int m0 = (wave & 3) * 32;
  const int n0 = (wave >> 2) * 64;
  const int blockN = blockIdx.x;                     // 0..249
  const int blockM = blockIdx.y;                     // 0..31

  FragC acc[2][4];
#pragma unroll
  for (int mi = 0; mi < 2; ++mi)
#pragma unroll
    for (int ni = 0; ni < 4; ++ni)
#pragma unroll
      for (int v = 0; v < 8; ++v) acc[mi][ni].f[v] = 0.0f;

  const int ar  = tid >> 1;
  const int akh = (tid & 1) * 16;
  const int bk  = tid >> 3;
  const int bn  = (tid & 7) * 16;
  const size_t arow = (size_t)(blockM * TM + ar);

  // Per-thread LDS byte address for async A staging. The flat address of a
  // __shared__ object is (aperture<<32)|lds_offset, so the low 32 bits are
  // the LDS byte address the async unit needs (ISA 10).
  const unsigned ldsA = (unsigned)(uintptr_t)&As[ar * LDSSA2 + akh];

  for (int kt = 0; kt < HIDDEN; kt += TK) {
    { // stage A: h already bf16 -> async global->LDS copy (32B per thread),
      // tracked by ASYNCcnt; offset: applies to BOTH lds and global address.
      const unsigned gOff =
          (unsigned)((arow * HIDDEN + (size_t)kt + akh) * sizeof(unsigned short));
      asm volatile(
          "global_load_async_to_lds_b128 %0, %1, %2\n\t"
          "global_load_async_to_lds_b128 %0, %1, %2 offset:16"
          :
          : "v"(ldsA), "v"(gOff), "s"(hbf)
          : "memory");
    }
    { // stage B: Why f32 -> bf16, transposed in LDS
      const float* src = Why + (size_t)(kt + bk) * VOCAB + blockN * TN + bn;
#pragma unroll
      for (int i = 0; i < 16; i += 4) {
        float4 f = *(const float4*)(src + i);
        Bs[(bn + i + 0) * LDSS + bk] = f32_to_bf16_rne(f.x);
        Bs[(bn + i + 1) * LDSS + bk] = f32_to_bf16_rne(f.y);
        Bs[(bn + i + 2) * LDSS + bk] = f32_to_bf16_rne(f.z);
        Bs[(bn + i + 3) * LDSS + bk] = f32_to_bf16_rne(f.w);
      }
    }
    asm volatile("s_wait_asynccnt 0" ::: "memory");  // A tile landed in LDS
    __syncthreads();

    FragAB a[2];
    const int mrow = lane & 15;
#pragma unroll
    for (int mi = 0; mi < 2; ++mi)
      load_a_frag(As, LDSSA2, m0 + mi * 16 + mrow, lane, a[mi]);
#pragma unroll
    for (int ni = 0; ni < 4; ++ni) {
      FragAB b;
      load_b_frag(Bs, n0 + ni * 16 + mrow, lane, b);
#pragma unroll
      for (int mi = 0; mi < 2; ++mi)
        acc[mi][ni].v = __builtin_amdgcn_wmma_f32_16x16x32_bf16(
            false, a[mi].v, false, b.v, (short)0, acc[mi][ni].v, false, false);
    }
    __syncthreads();
  }

  const int nlane = lane & 15;
  const int mhalf = (lane >> 4) ? 8 : 0;
#pragma unroll
  for (int mi = 0; mi < 2; ++mi)
#pragma unroll
    for (int ni = 0; ni < 4; ++ni) {
      const int col = blockN * TN + n0 + ni * 16 + nlane;
      const float byv = by[col];
#pragma unroll
      for (int v = 0; v < 8; ++v) {
        const int row = blockM * TM + m0 + mi * 16 + mhalf + v;
        out[(size_t)row * VOCAB + col] = acc[mi][ni].f[v] + byv;
      }
    }
}

// ---------------------------------------------------------------------------
extern "C" void kernel_launch(void* const* d_in, const int* in_sizes, int n_in,
                              void* d_out, int out_size, void* d_ws, size_t ws_size,
                              hipStream_t stream) {
  (void)in_sizes; (void)n_in; (void)out_size; (void)ws_size;

  const float* hprev = (const float*)d_in[0];   // [1, 2048]
  const float* x     = (const float*)d_in[1];   // [4096, 32000]
  const float* Wxh   = (const float*)d_in[2];   // [32000, 2048]
  const float* Whh   = (const float*)d_in[3];   // [2048, 2048]
  const float* Why   = (const float*)d_in[4];   // [2048, 32000]
  const float* bh    = (const float*)d_in[5];   // [2048]
  const float* by    = (const float*)d_in[6];   // [32000]
  float* out = (float*)d_out;                   // logits ++ h_last

  // workspace: hc (2048 f32 = 8KB) | h in bf16 (4096*2048*2 = 16.8MB)
  float* hc = (float*)d_ws;
  unsigned short* hbf = (unsigned short*)((char*)d_ws + 8192);

  hcontrib_kernel<<<HIDDEN / 256, 256, 0, stream>>>(hprev, Whh, bh, hc);

  gemm1_tanh_kernel<<<dim3(HIDDEN / TN, TSEQ / TM), 256, 0, stream>>>(
      x, Wxh, hc, hbf, out + (size_t)TSEQ * VOCAB);

  gemm2_kernel<<<dim3(VOCAB / TN, TSEQ / TM), 256, 0, stream>>>(
      hbf, Why, by, out);
}
